// SparseQuery_25013889532676
// MI455X (gfx1250) — matrix operand order repaired and checked by reading, not compiled, not run on hardware
//
#include <hip/hip_runtime.h>
#include <hip/hip_bf16.h>

// Problem constants
#define I_FEAT 1024
#define NHEADS 16
#define HDIM   64
#define HSIZE  512
#define GSIZE  128
#define TOK    8192   // 4 * 2048 tokens
#define MTOK   32     // tokens per expert block (2 WMMA M-tiles)

typedef __attribute__((ext_vector_type(16))) __bf16 v16bf;
typedef __attribute__((ext_vector_type(8)))  __bf16 v8bf;
typedef __attribute__((ext_vector_type(8)))  float  v8f;
typedef int v4i __attribute__((ext_vector_type(4)));

// Async global->LDS staging (gfx1250 ASYNCcnt path), guarded so a missing
// builtin falls back to plain LDS copies instead of breaking the build.
#if defined(__has_builtin)
#if __has_builtin(__builtin_amdgcn_global_load_async_to_lds_b128)
#define HAVE_ASYNC_LDS 1
#endif
#if __has_builtin(__builtin_amdgcn_s_wait_asynccnt)
#define HAVE_WAIT_ASYNC_BUILTIN 1
#endif
#endif

__device__ __forceinline__ void wait_asynccnt0() {
#if defined(HAVE_WAIT_ASYNC_BUILTIN)
    __builtin_amdgcn_s_wait_asynccnt(0);
#else
    asm volatile("s_wait_asynccnt 0x0" ::: "memory");
#endif
}

__device__ __forceinline__ float gelu_exact(float x) {
    return 0.5f * x * (1.0f + erff(x * 0.70710678118654752f));
}

__device__ __forceinline__ v16bf join16(v8bf lo, v8bf hi) {
    v16bf r;
#pragma unroll
    for (int i = 0; i < 8; ++i) { r[i] = lo[i]; r[i + 8] = hi[i]; }
    return r;
}

__device__ __forceinline__ v8f wmma_bf16(v16bf a, v16bf b, v8f c) {
    return __builtin_amdgcn_wmma_f32_16x16x32_bf16(false, a, false, b, (short)0, c,
                                                   false, false);
}

// ---------------------------------------------------------------------------
// Re-layout input_experts [16][1024][512] fp32 -> bf16 WMMA-B-fragment order:
//   dst[(((h*32 + ks)*32 + nt)*32 + lane)*16 + i]  holds B[k][n] with
//   k = ks*32 + (i>>3)*16 + (lane>>4)*8 + (i&7),  n = nt*16 + (lane&15)
// ---------------------------------------------------------------------------
__global__ void relayout_win_kernel(const float* __restrict__ w, __bf16* __restrict__ dst) {
    const long long total = (long long)NHEADS * I_FEAT * HSIZE;  // 2^23
    for (long long idx = (long long)blockIdx.x * blockDim.x + threadIdx.x; idx < total;
         idx += (long long)gridDim.x * blockDim.x) {
        int i    = (int)(idx & 15);
        int lane = (int)((idx >> 4) & 31);
        int nt   = (int)((idx >> 9) & 31);   // H/16 = 32 tiles
        int ks   = (int)((idx >> 14) & 31);  // I/32 = 32 k-steps
        int h    = (int)(idx >> 19);
        int k = ks * 32 + (i >> 3) * 16 + (lane >> 4) * 8 + (i & 7);
        int n = nt * 16 + (lane & 15);
        dst[idx] = (__bf16)w[((long long)h * I_FEAT + k) * HSIZE + n];
    }
}

// output_experts [16][512][64] fp32 -> fragment order (H/32=16 ksteps, D/16=4 ntiles)
__global__ void relayout_wout_kernel(const float* __restrict__ w, __bf16* __restrict__ dst) {
    const long long total = (long long)NHEADS * HSIZE * HDIM;    // 2^19
    for (long long idx = (long long)blockIdx.x * blockDim.x + threadIdx.x; idx < total;
         idx += (long long)gridDim.x * blockDim.x) {
        int i    = (int)(idx & 15);
        int lane = (int)((idx >> 4) & 31);
        int nt   = (int)((idx >> 9) & 3);
        int ks   = (int)((idx >> 11) & 15);
        int h    = (int)(idx >> 15);
        int k = ks * 32 + (i >> 3) * 16 + (lane >> 4) * 8 + (i & 7);
        int n = nt * 16 + (lane & 15);
        dst[idx] = (__bf16)w[((long long)h * HSIZE + k) * HDIM + n];
    }
}

// Normalize centroids; fold exp(temperature)/sqrt(G) into one scale.
__global__ void prep_router_kernel(const float* __restrict__ cent,
                                   const float* __restrict__ temp,
                                   float* __restrict__ c_norm,
                                   float* __restrict__ scale) {
    int t = threadIdx.x;
    if (t < NHEADS) {
        float ss = 0.0f;
        for (int g = 0; g < GSIZE; ++g) { float v = cent[t * GSIZE + g]; ss += v * v; }
        float inv = 1.0f / fmaxf(sqrtf(ss), 1e-12f);
        for (int g = 0; g < GSIZE; ++g) c_norm[t * GSIZE + g] = cent[t * GSIZE + g] * inv;
    }
    if (t == 0) scale[0] = expf(temp[0]) * rsqrtf((float)GSIZE);
}

// ---------------------------------------------------------------------------
// Router: one wave (32 lanes) per 16-token tile.
//   z = x * w_router^T via bf16 WMMA (16x16x32), fp32 softmax/top-2,
//   compacts (token, prob) into per-head buckets; also emits x in bf16.
// ---------------------------------------------------------------------------
__global__ void __launch_bounds__(32) router_kernel(
    const float* __restrict__ x, const float* __restrict__ w_router,
    const float* __restrict__ c_norm, const float* __restrict__ scale_p,
    __bf16* __restrict__ x_bf, int* __restrict__ counts,
    int* __restrict__ btok, float* __restrict__ bprob) {
    __shared__ float zl[16][GSIZE];

    const int lane = threadIdx.x & 31;
    const int half = lane >> 4;
    const int row  = lane & 15;
    const int tok0 = blockIdx.x * 16;

    v8f acc[8];
#pragma unroll
    for (int t = 0; t < 8; ++t)
#pragma unroll
        for (int j = 0; j < 8; ++j) acc[t][j] = 0.0f;

    const float* xrow  = x    + (size_t)(tok0 + row) * I_FEAT;
    __bf16*      xbrow = x_bf + (size_t)(tok0 + row) * I_FEAT;

    for (int kc = 0; kc < I_FEAT / 32; ++kc) {
        const int kb1 = kc * 32 + half * 16;    // first 8-wide run of this lane
        const int kb2 = kb1 + 8;                // second run (contiguous 16 k per lane half)
        float tmp[16];
        *(float4*)(tmp + 0)  = *(const float4*)(xrow + kb1);
        *(float4*)(tmp + 4)  = *(const float4*)(xrow + kb1 + 4);
        *(float4*)(tmp + 8)  = *(const float4*)(xrow + kb2);
        *(float4*)(tmp + 12) = *(const float4*)(xrow + kb2 + 4);
        v16bf afrag;
        v8bf lo, hi;
#pragma unroll
        for (int i = 0; i < 8; ++i) {
            lo[i] = (__bf16)tmp[i];
            hi[i] = (__bf16)tmp[8 + i];
            afrag[i] = lo[i];
            afrag[8 + i] = hi[i];
        }
        *(v8bf*)(xbrow + kb1) = lo;             // persist bf16 x for expert pass
        *(v8bf*)(xbrow + kb2) = hi;

#pragma unroll
        for (int gt = 0; gt < 8; ++gt) {        // 8 gate tiles of 16 -> G=128
            // B[k][n] = w_router[n][k]; each lane reads 16 consecutive k
            const float* wr = w_router + (size_t)(gt * 16 + row) * I_FEAT + kc * 32 + half * 16;
            float bt[16];
            *(float4*)(bt + 0)  = ((const float4*)wr)[0];
            *(float4*)(bt + 4)  = ((const float4*)wr)[1];
            *(float4*)(bt + 8)  = ((const float4*)wr)[2];
            *(float4*)(bt + 12) = ((const float4*)wr)[3];
            v16bf bfrag;
#pragma unroll
            for (int i = 0; i < 16; ++i) bfrag[i] = (__bf16)bt[i];
            acc[gt] = wmma_bf16(afrag, bfrag, acc[gt]);
        }
    }

    // C layout: lane,l -> row M = l + 8*half, col N = lane&15
#pragma unroll
    for (int gt = 0; gt < 8; ++gt)
#pragma unroll
        for (int v = 0; v < 8; ++v) zl[v + 8 * half][gt * 16 + row] = acc[gt][v];
    __syncthreads();

    const float sc = scale_p[0];
    if (lane < 16) {
        const int t = lane;
        float ss = 0.0f;
        for (int g = 0; g < GSIZE; ++g) { float v = zl[t][g]; ss += v * v; }
        const float invn = 1.0f / fmaxf(sqrtf(ss), 1e-12f);

        float lg[NHEADS];
        float m = -1e30f;
        for (int n = 0; n < NHEADS; ++n) {
            float d = 0.0f;
            for (int g = 0; g < GSIZE; ++g) d += zl[t][g] * c_norm[n * GSIZE + g];
            lg[n] = d * invn * sc;
            m = fmaxf(m, lg[n]);
        }
        float s = 0.0f;
        for (int n = 0; n < NHEADS; ++n) { lg[n] = expf(lg[n] - m); s += lg[n]; }
        const float invs = 1.0f / s;

        int i1 = 0; float p1 = -1.0f;
        for (int n = 0; n < NHEADS; ++n) {
            float p = lg[n] * invs;
            if (p > p1) { p1 = p; i1 = n; }
        }
        int i2 = 0; float p2 = -1.0f;
        for (int n = 0; n < NHEADS; ++n) {
            if (n == i1) continue;
            float p = lg[n] * invs;
            if (p > p2) { p2 = p; i2 = n; }
        }
        const int tokid = tok0 + t;
        int pos = atomicAdd(&counts[i1], 1);
        btok[i1 * TOK + pos] = tokid; bprob[i1 * TOK + pos] = p1;
        pos = atomicAdd(&counts[i2], 1);
        btok[i2 * TOK + pos] = tokid; bprob[i2 * TOK + pos] = p2;
    }
}

// ---------------------------------------------------------------------------
// Grouped-GEMM expert pass: block = (head, 32-token tile), 4 waves.
//   2 M-tiles per block -> each B fragment feeds 2 WMMAs (halves weight traffic).
//   hidden(32x512) = x_tile(32x1024) @ Win[h]; GELU;
//   out(32x64) = hidden @ Wout[h] with K split across waves + LDS reduction.
//   x tile staged via GLOBAL_LOAD_ASYNC_TO_LDS (ASYNCcnt) when available.
// ---------------------------------------------------------------------------
__global__ void __launch_bounds__(128) expert_kernel(
    const __bf16* __restrict__ x_bf, const __bf16* __restrict__ w_in,
    const __bf16* __restrict__ w_out, const int* __restrict__ counts,
    const int* __restrict__ btok, const float* __restrict__ bprob,
    float* __restrict__ out) {
    const int h    = blockIdx.y;
    const int cnt  = counts[h];
    const int base = blockIdx.x * MTOK;
    if (base >= cnt) return;
    const int nvalid = min(MTOK, cnt - base);

    __shared__ __bf16 xs[MTOK][64];                  // 4 KB staging tile
    __shared__ char   smem[MTOK * HSIZE * 2];        // 32 KB: hid (bf16) then part (f32)
    __shared__ int    stok[MTOK];
    __shared__ float  sprob[MTOK];
    __bf16 (*hid)[HSIZE]       = (__bf16(*)[HSIZE])smem;
    float  (*part)[MTOK][HDIM] = (float(*)[MTOK][HDIM])smem;  // 4*32*64*4 = 32 KB

    const int tid = threadIdx.x;
    if (tid < MTOK) {
        const int idx = base + tid;
        if (idx < cnt) { stok[tid] = btok[h * TOK + idx]; sprob[tid] = bprob[h * TOK + idx]; }
        else           { stok[tid] = btok[h * TOK + base]; sprob[tid] = 0.0f; }
    }
    __syncthreads();

    const int wave = tid >> 5;
    const int lane = tid & 31;
    const int half = lane >> 4;
    const int col  = lane & 15;

    // cooperative x staging: 32 rows x 64 cols bf16 = 256 x 16B chunks, 2/thread
    const int r0 = tid >> 3,          s0 = tid & 7;          // chunk tid
    const int r1 = (tid + 128) >> 3,  s1 = (tid + 128) & 7;  // chunk tid+128
    const __bf16* g0 = x_bf + (size_t)stok[r0] * I_FEAT + s0 * 8;
    const __bf16* g1 = x_bf + (size_t)stok[r1] * I_FEAT + s1 * 8;

    v8f acc[2][8];
#pragma unroll
    for (int m = 0; m < 2; ++m)
#pragma unroll
        for (int t = 0; t < 8; ++t)
#pragma unroll
            for (int j = 0; j < 8; ++j) acc[m][t][j] = 0.0f;

    for (int kc = 0; kc < I_FEAT / 64; ++kc) {
        __syncthreads();
#if defined(HAVE_ASYNC_LDS)
        __builtin_amdgcn_global_load_async_to_lds_b128(
            (v4i*)(void*)(g0 + kc * 64), (v4i*)(void*)(&xs[r0][s0 * 8]), 0, 0);
        __builtin_amdgcn_global_load_async_to_lds_b128(
            (v4i*)(void*)(g1 + kc * 64), (v4i*)(void*)(&xs[r1][s1 * 8]), 0, 0);
        wait_asynccnt0();
#else
        *(v8bf*)(&xs[r0][s0 * 8]) = *(const v8bf*)(g0 + kc * 64);
        *(v8bf*)(&xs[r1][s1 * 8]) = *(const v8bf*)(g1 + kc * 64);
#endif
        __syncthreads();
#pragma unroll
        for (int s = 0; s < 2; ++s) {
            const int ks = kc * 2 + s;
            const int kb = s * 32 + half * 8;
            v16bf a0 = join16(*(const v8bf*)(&xs[col][kb]),
                              *(const v8bf*)(&xs[col][kb + 16]));
            v16bf a1 = join16(*(const v8bf*)(&xs[16 + col][kb]),
                              *(const v8bf*)(&xs[16 + col][kb + 16]));
#pragma unroll
            for (int nt = 0; nt < 8; ++nt) {
                const int ntg = wave * 8 + nt;  // this wave's H range
                const __bf16* bp =
                    w_in + ((((size_t)h * 32 + ks) * 32 + ntg) * 32 + lane) * 16;
                v16bf bfrag = *(const v16bf*)bp;     // one 32B fragment, 2 WMMAs
                acc[0][nt] = wmma_bf16(a0, bfrag, acc[0][nt]);
                acc[1][nt] = wmma_bf16(a1, bfrag, acc[1][nt]);
            }
        }
    }

    // GELU, stash hidden as bf16 (plain [row][H] layout)
#pragma unroll
    for (int m = 0; m < 2; ++m)
#pragma unroll
        for (int nt = 0; nt < 8; ++nt) {
            const int n = wave * 128 + nt * 16 + col;
#pragma unroll
            for (int v = 0; v < 8; ++v)
                hid[m * 16 + v + 8 * half][n] = (__bf16)gelu_exact(acc[m][nt][v]);
        }
    __syncthreads();

    // GEMM2: K=512 split 128 per wave; 4 N-tiles cover D=64, 2 M-tiles
    v8f oacc[2][4];
#pragma unroll
    for (int m = 0; m < 2; ++m)
#pragma unroll
        for (int t = 0; t < 4; ++t)
#pragma unroll
            for (int j = 0; j < 8; ++j) oacc[m][t][j] = 0.0f;

#pragma unroll
    for (int s = 0; s < 4; ++s) {
        const int kb = wave * 128 + s * 32 + half * 8;
        v16bf a0 = join16(*(const v8bf*)(&hid[col][kb]),
                          *(const v8bf*)(&hid[col][kb + 16]));
        v16bf a1 = join16(*(const v8bf*)(&hid[16 + col][kb]),
                          *(const v8bf*)(&hid[16 + col][kb + 16]));
        const int ks = wave * 4 + s;
#pragma unroll
        for (int nt = 0; nt < 4; ++nt) {
            const __bf16* bp =
                w_out + ((((size_t)h * 16 + ks) * 4 + nt) * 32 + lane) * 16;
            v16bf bfrag = *(const v16bf*)bp;
            oacc[0][nt] = wmma_bf16(a0, bfrag, oacc[0][nt]);
            oacc[1][nt] = wmma_bf16(a1, bfrag, oacc[1][nt]);
        }
    }
    __syncthreads();   // all waves done reading hid; smem now becomes `part`

#pragma unroll
    for (int m = 0; m < 2; ++m)
#pragma unroll
        for (int nt = 0; nt < 4; ++nt)
#pragma unroll
            for (int v = 0; v < 8; ++v)
                part[wave][m * 16 + v + 8 * half][nt * 16 + col] = oacc[m][nt][v];
    __syncthreads();

    // cross-wave reduce, gate, scatter: thread -> (row M, 16-wide col chunk)
    const int M  = tid >> 2;
    const int db = (tid & 3) * 16;
    if (M < nvalid) {
        const float p = sprob[M];
        float* op = out + (size_t)stok[M] * (NHEADS * HDIM) + h * HDIM + db;
#pragma unroll
        for (int j = 0; j < 16; ++j) {
            const int d = db + j;
            op[j] = p * (part[0][M][d] + part[1][M][d] + part[2][M][d] + part[3][M][d]);
        }
    }
}

// ---------------------------------------------------------------------------
extern "C" void kernel_launch(void* const* d_in, const int* in_sizes, int n_in,
                              void* d_out, int out_size, void* d_ws, size_t ws_size,
                              hipStream_t stream) {
    (void)in_sizes; (void)n_in; (void)ws_size;
    const float* x      = (const float*)d_in[0];
    const float* w_rt   = (const float*)d_in[1];
    const float* cent   = (const float*)d_in[2];
    const float* temp   = (const float*)d_in[3];
    const float* w_inf  = (const float*)d_in[4];
    const float* w_outf = (const float*)d_in[5];
    float* out = (float*)d_out;

    char* ws = (char*)d_ws;
    size_t off = 0;
    auto carve = [&](size_t bytes) -> char* {
        off = (off + 255) & ~(size_t)255;
        char* p = ws + off;
        off += bytes;
        return p;
    };
    __bf16* x_bf   = (__bf16*)carve((size_t)TOK * I_FEAT * 2);
    __bf16* win_b  = (__bf16*)carve((size_t)NHEADS * I_FEAT * HSIZE * 2);
    __bf16* wout_b = (__bf16*)carve((size_t)NHEADS * HSIZE * HDIM * 2);
    float*  c_norm = (float*)carve((size_t)NHEADS * GSIZE * 4);
    float*  scale  = (float*)carve(256);
    int*    counts = (int*)carve(256);
    int*    btok   = (int*)carve((size_t)NHEADS * TOK * 4);
    float*  bprob  = (float*)carve((size_t)NHEADS * TOK * 4);

    (void)hipMemsetAsync(d_out, 0, (size_t)out_size * sizeof(float), stream);
    (void)hipMemsetAsync(counts, 0, 256, stream);

    relayout_win_kernel<<<4096, 256, 0, stream>>>(w_inf, win_b);
    relayout_wout_kernel<<<512, 256, 0, stream>>>(w_outf, wout_b);
    prep_router_kernel<<<1, 32, 0, stream>>>(cent, temp, c_norm, scale);
    router_kernel<<<TOK / 16, 32, 0, stream>>>(x, w_rt, c_norm, scale, x_bf, counts,
                                               btok, bprob);
    expert_kernel<<<dim3(TOK / MTOK, NHEADS), 128, 0, stream>>>(x_bf, win_b, wout_b,
                                                                counts, btok, bprob, out);
}
